// RelMultiHeadAttention_56908316672218
// MI455X (gfx1250) — matrix operand order repaired and checked by reading, not compiled, not run on hardware
//
#include <hip/hip_runtime.h>

typedef __attribute__((ext_vector_type(16))) __bf16 v16bf;
typedef __attribute__((ext_vector_type(8)))  float  v8f;

#define N_HEAD 16
#define D_HEAD 64
#define EMBED  1024
#define BATCH  2
#define QLEN   1024
#define MEMLEN 1024
#define KLEN   2048
#define GSTR   40   // LDS stride for 64x32 GEMM tiles (bf16 elems)
#define SSTR   65   // f32 S tile stride (conflict-free row access)
#define TSTR   97   // f32 BD-band result stride

union frag16 { v16bf v; uint4 q[2]; };

__device__ __forceinline__ v8f vzero8() {
  v8f z;
#pragma unroll
  for (int i = 0; i < 8; ++i) z[i] = 0.0f;
  return z;
}

// A-matrix fragment (16 rows x K=32), row-major LDS tile, stride in elements.
// ISA 7.12.2: lane<16 -> elems 0..7 = K0..7, 8..15 = K16..23 ; lane>=16 -> +8.
__device__ __forceinline__ v16bf load_frag_a(const __bf16* tile, int stride, int lane) {
  const int r  = lane & 15;
  const int hs = (lane >> 4) << 3;
  const __bf16* p = tile + r * stride + hs;
  frag16 u;
  u.q[0] = *reinterpret_cast<const uint4*>(p);
  u.q[1] = *reinterpret_cast<const uint4*>(p + 16);
  return u.v;
}

// B-matrix fragment (K=32 x 16 cols). Tile stored row-major as B^T: [n][k].
// lane<16 -> elems 0..7 = K0..7, 8..15 = K16..23 ; lane>=16 -> +8 (matches A).
__device__ __forceinline__ v16bf load_frag_b(const __bf16* tile, int stride, int lane) {
  const int n  = lane & 15;
  const int k0 = (lane >> 4) << 3;
  const __bf16* p = tile + n * stride + k0;
  frag16 u;
  u.q[0] = *reinterpret_cast<const uint4*>(p);
  u.q[1] = *reinterpret_cast<const uint4*>(p + 16);
  return u.v;
}

__device__ __forceinline__ v8f wmma_bf16(v16bf a, v16bf b, v8f c) {
  return __builtin_amdgcn_wmma_f32_16x16x32_bf16(false, a, false, b, (short)0, c,
                                                 false, false);
}

__device__ __forceinline__ void copy32bf(__bf16* dst, const __bf16* src) {
  const uint4* s = reinterpret_cast<const uint4*>(src);
  uint4* d = reinterpret_cast<uint4*>(dst);
#pragma unroll
  for (int t = 0; t < 4; ++t) d[t] = s[t];
}
__device__ __forceinline__ void zero64bf(__bf16* dst) {
  uint4 z; z.x = 0u; z.y = 0u; z.z = 0u; z.w = 0u;
  uint4* d = reinterpret_cast<uint4*>(dst);
#pragma unroll
  for (int t = 0; t < 8; ++t) d[t] = z;
}

// ---- CDNA5 async global->LDS (CDNA5 ISA sec.4, ASYNCcnt) ------
// Per-lane: LDS[vdst_lds_addr] <= MEM[vaddr], 16 bytes, tracked by ASYNCcnt.
// LDS aperture maps addr[31:0] directly to the LDS offset (ISA 10.2), so the
// low 32 bits of a generic (flat) shared pointer are a valid LDS address.
__device__ __forceinline__ unsigned lds_off_of(const void* p) {
  return (unsigned)(size_t)p;
}
__device__ __forceinline__ void async_copy16(unsigned lds_addr, const void* gptr) {
  asm volatile("global_load_async_to_lds_b128 %0, %1, off"
               :: "v"(lds_addr), "v"(gptr)
               : "memory");
}
// 32 bf16 (64B) per call site, 4 async b128 per lane
__device__ __forceinline__ void async_copy32bf(__bf16* dst, const __bf16* src) {
  unsigned base = lds_off_of(dst);
#pragma unroll
  for (int t = 0; t < 4; ++t) async_copy16(base + 16u * t, src + 8 * t);
}
__device__ __forceinline__ void async_copy64bf(__bf16* dst, const __bf16* src) {
  unsigned base = lds_off_of(dst);
#pragma unroll
  for (int t = 0; t < 8; ++t) async_copy16(base + 16u * t, src + 8 * t);
}
__device__ __forceinline__ void async_wait0() {
  asm volatile("s_wait_asynccnt 0x0" ::: "memory");
}

// ---------------------------------------------------------------- projections
// q = w @ Wq ; emits q + r_w_bias and q + r_r_bias as bf16 in [B,H,Q,64]
__global__ __launch_bounds__(128)
void proj_q_kernel(const float* __restrict__ w, const float* __restrict__ Wq,
                   const float* __restrict__ rwb, const float* __restrict__ rrb,
                   __bf16* __restrict__ qrw, __bf16* __restrict__ qrr) {
  __shared__ __bf16 sA[64 * GSTR];
  __shared__ __bf16 sW[64 * GSTR];
  const int m0 = blockIdx.x * 64, n0 = blockIdx.y * 64;
  const int tid = threadIdx.x, wid = tid >> 5, lane = tid & 31;
  v8f acc[4];
#pragma unroll
  for (int cb = 0; cb < 4; ++cb) acc[cb] = vzero8();
  for (int k0 = 0; k0 < EMBED; k0 += 32) {
    for (int e = tid; e < 64 * 32; e += 128) {
      int rr_ = e >> 5, cc = e & 31;
      sA[rr_ * GSTR + cc] = (__bf16)w[(size_t)(m0 + rr_) * EMBED + k0 + cc];
    }
    for (int e = tid; e < 64 * 32; e += 128) {
      int kk = e >> 6, nn = e & 63;
      sW[nn * GSTR + kk] = (__bf16)Wq[(size_t)(k0 + kk) * EMBED + n0 + nn];
    }
    __syncthreads();
    v16bf a = load_frag_a(sA + wid * 16 * GSTR, GSTR, lane);
#pragma unroll
    for (int cb = 0; cb < 4; ++cb)
      acc[cb] = wmma_bf16(a, load_frag_b(sW + cb * 16 * GSTR, GSTR, lane), acc[cb]);
    __syncthreads();
  }
  const int h = n0 >> 6, b = m0 >> 10, ib = m0 & 1023;
  const int n = lane & 15, hi = lane >> 4;
#pragma unroll
  for (int cb = 0; cb < 4; ++cb)
#pragma unroll
    for (int v = 0; v < 8; ++v) {
      int row = wid * 16 + v + 8 * hi, d = cb * 16 + n;
      size_t gi = ((size_t)(b * N_HEAD + h) * QLEN + (ib + row)) * D_HEAD + d;
      float val = acc[cb][v];
      qrw[gi] = (__bf16)(val + rwb[h * D_HEAD + d]);
      qrr[gi] = (__bf16)(val + rrb[h * D_HEAD + d]);
    }
}

// k/v = concat(member, w) @ W ; bf16 out in [B,H,K,64]
__global__ __launch_bounds__(128)
void proj_cat_kernel(const float* __restrict__ member, const float* __restrict__ w,
                     const float* __restrict__ W, __bf16* __restrict__ outp) {
  __shared__ __bf16 sA[64 * GSTR];
  __shared__ __bf16 sW[64 * GSTR];
  const int m0 = blockIdx.x * 64, n0 = blockIdx.y * 64;
  const int tid = threadIdx.x, wid = tid >> 5, lane = tid & 31;
  const int b = m0 >> 11, kkb = m0 & 2047;
  v8f acc[4];
#pragma unroll
  for (int cb = 0; cb < 4; ++cb) acc[cb] = vzero8();
  for (int k0 = 0; k0 < EMBED; k0 += 32) {
    for (int e = tid; e < 64 * 32; e += 128) {
      int rr_ = e >> 5, cc = e & 31;
      int kk = kkb + rr_;
      const float* src = (kk < MEMLEN)
          ? (member + ((size_t)b * MEMLEN + kk) * EMBED)
          : (w + ((size_t)b * QLEN + (kk - MEMLEN)) * EMBED);
      sA[rr_ * GSTR + cc] = (__bf16)src[k0 + cc];
    }
    for (int e = tid; e < 64 * 32; e += 128) {
      int kk = e >> 6, nn = e & 63;
      sW[nn * GSTR + kk] = (__bf16)W[(size_t)(k0 + kk) * EMBED + n0 + nn];
    }
    __syncthreads();
    v16bf a = load_frag_a(sA + wid * 16 * GSTR, GSTR, lane);
#pragma unroll
    for (int cb = 0; cb < 4; ++cb)
      acc[cb] = wmma_bf16(a, load_frag_b(sW + cb * 16 * GSTR, GSTR, lane), acc[cb]);
    __syncthreads();
  }
  const int h = n0 >> 6;
  const int n = lane & 15, hi = lane >> 4;
#pragma unroll
  for (int cb = 0; cb < 4; ++cb)
#pragma unroll
    for (int v = 0; v < 8; ++v) {
      int row = wid * 16 + v + 8 * hi, d = cb * 16 + n;
      outp[((size_t)(b * N_HEAD + h) * KLEN + (kkb + row)) * D_HEAD + d] =
          (__bf16)acc[cb][v];
    }
}

// rr = r @ Wr ; bf16 out in [H,K,64]
__global__ __launch_bounds__(128)
void proj_r_kernel(const float* __restrict__ r, const float* __restrict__ Wr,
                   __bf16* __restrict__ outp) {
  __shared__ __bf16 sA[64 * GSTR];
  __shared__ __bf16 sW[64 * GSTR];
  const int m0 = blockIdx.x * 64, n0 = blockIdx.y * 64;
  const int tid = threadIdx.x, wid = tid >> 5, lane = tid & 31;
  v8f acc[4];
#pragma unroll
  for (int cb = 0; cb < 4; ++cb) acc[cb] = vzero8();
  for (int k0 = 0; k0 < EMBED; k0 += 32) {
    for (int e = tid; e < 64 * 32; e += 128) {
      int rr_ = e >> 5, cc = e & 31;
      sA[rr_ * GSTR + cc] = (__bf16)r[(size_t)(m0 + rr_) * EMBED + k0 + cc];
    }
    for (int e = tid; e < 64 * 32; e += 128) {
      int kk = e >> 6, nn = e & 63;
      sW[nn * GSTR + kk] = (__bf16)Wr[(size_t)(k0 + kk) * EMBED + n0 + nn];
    }
    __syncthreads();
    v16bf a = load_frag_a(sA + wid * 16 * GSTR, GSTR, lane);
#pragma unroll
    for (int cb = 0; cb < 4; ++cb)
      acc[cb] = wmma_bf16(a, load_frag_b(sW + cb * 16 * GSTR, GSTR, lane), acc[cb]);
    __syncthreads();
  }
  const int h = n0 >> 6;
  const int n = lane & 15, hi = lane >> 4;
#pragma unroll
  for (int cb = 0; cb < 4; ++cb)
#pragma unroll
    for (int v = 0; v < 8; ++v) {
      int row = wid * 16 + v + 8 * hi, d = cb * 16 + n;
      outp[((size_t)h * KLEN + (m0 + row)) * D_HEAD + d] = (__bf16)acc[cb][v];
    }
}

// ------------------------------------------------------------ flash attention
// q-tile = 32 rows, k-tile = 64 cols. S = (q+rwb)K^T ; BD via band of r@Wr
// rows c = j + 1023 - i (rel_shift identity); online softmax; O += P V.
// K-tile / band / Q-tile copies use CDNA5 async global->LDS (ASYNCcnt).
__global__ __launch_bounds__(128)
void attn_kernel(const __bf16* __restrict__ qrw, const __bf16* __restrict__ qrr,
                 const __bf16* __restrict__ kb, const __bf16* __restrict__ vb,
                 const __bf16* __restrict__ rb, __bf16* __restrict__ aout) {
  __shared__ __bf16 sQW[32 * 64];
  __shared__ __bf16 sQR[32 * 64];
  __shared__ __bf16 sK[64 * 64];
  __shared__ __bf16 sVT[64 * 64];     // V^T: [d][kk]
  __shared__ __bf16 sBand[96 * 64];   // rr rows c0 .. c0+95
  __shared__ __bf16 sP[32 * 64];
  __shared__ float  sS[32 * SSTR];
  __shared__ float  sT[32 * TSTR];    // BD band result [32][96]
  __shared__ float  sM[32], sL[32], sAl[32];

  const int b = blockIdx.z, h = blockIdx.y, qt = blockIdx.x;
  const int i0 = qt * 32;
  const int tid = threadIdx.x, wid = tid >> 5, lane = tid & 31;
  const int rs = wid & 1;   // row strip (q rows rs*16 ..)
  const int ch = wid >> 1;  // column half

  const __bf16* qwg = qrw + ((size_t)(b * N_HEAD + h) * QLEN + i0) * D_HEAD;
  const __bf16* qrg = qrr + ((size_t)(b * N_HEAD + h) * QLEN + i0) * D_HEAD;
  const __bf16* kg  = kb + (size_t)(b * N_HEAD + h) * KLEN * D_HEAD;
  const __bf16* vg  = vb + (size_t)(b * N_HEAD + h) * KLEN * D_HEAD;
  const __bf16* rg  = rb + (size_t)h * KLEN * D_HEAD;

  if (tid < 64) {
    int rr_ = tid >> 1, half = tid & 1;
    async_copy32bf(sQW + rr_ * 64 + half * 32, qwg + rr_ * 64 + half * 32);
    async_copy32bf(sQR + rr_ * 64 + half * 32, qrg + rr_ * 64 + half * 32);
  }
  if (tid < 32) { sM[tid] = -3.0e38f; sL[tid] = 0.0f; }
  async_wait0();
  __syncthreads();

  v16bf aqw[2], aqr[2];
#pragma unroll
  for (int kh = 0; kh < 2; ++kh) {
    aqw[kh] = load_frag_a(sQW + rs * 16 * 64 + kh * 32, 64, lane);
    aqr[kh] = load_frag_a(sQR + rs * 16 * 64 + kh * 32, 64, lane);
  }
  v8f o[2];
  o[0] = vzero8(); o[1] = vzero8();

  const int nkt = ((i0 + MEMLEN + 31) >> 6) + 1;
  const int n = lane & 15, hi = lane >> 4;

  for (int kt = 0; kt < nkt; ++kt) {
    const int j0 = kt * 64;
    __syncthreads();
    { // K tile (row-major copy, async to LDS)
      int rr_ = tid >> 1, half = tid & 1;
      async_copy32bf(sK + rr_ * 64 + half * 32, kg + (size_t)(j0 + rr_) * 64 + half * 32);
    }
    if (tid < 96) { // rel-position band, clamp past KLEN to zero
      int c = j0 - i0 + 992 + tid;
      if (c < KLEN) async_copy64bf(sBand + tid * 64, rg + (size_t)c * 64);
      else          zero64bf(sBand + tid * 64);
    }
    for (int e = tid; e < 64 * 64; e += 128) { // V transposed into LDS (VALU path)
      int kk = e >> 6, d = e & 63;
      sVT[d * 64 + kk] = vg[(size_t)(j0 + kk) * 64 + d];
    }
    if (kt + 1 < nkt) {
      __builtin_prefetch(kg + (size_t)(j0 + 64) * 64, 0, 1);
      __builtin_prefetch(vg + (size_t)(j0 + 64) * 64, 0, 1);
    }
    async_wait0();
    __syncthreads();

    // S (content scores): wave covers rows rs*16.. x cols ch*32..+31
#pragma unroll
    for (int cb = 0; cb < 2; ++cb) {
      v8f acc = vzero8();
#pragma unroll
      for (int kh = 0; kh < 2; ++kh)
        acc = wmma_bf16(aqw[kh],
                        load_frag_b(sK + (ch * 32 + cb * 16) * 64 + kh * 32, 64, lane),
                        acc);
#pragma unroll
      for (int v = 0; v < 8; ++v)
        sS[(rs * 16 + v + 8 * hi) * SSTR + ch * 32 + cb * 16 + n] = acc[v];
    }
    // T (BD band): 96 cols split 3 blocks per column-half
#pragma unroll
    for (int t = 0; t < 3; ++t) {
      const int cb2 = ch * 3 + t;
      v8f acc = vzero8();
#pragma unroll
      for (int kh = 0; kh < 2; ++kh)
        acc = wmma_bf16(aqr[kh],
                        load_frag_b(sBand + cb2 * 16 * 64 + kh * 32, 64, lane), acc);
#pragma unroll
      for (int v = 0; v < 8; ++v)
        sT[(rs * 16 + v + 8 * hi) * TSTR + cb2 * 16 + n] = acc[v];
    }
    __syncthreads();

    // online softmax (one thread per q row)
    if (tid < 32) {
      const int row = tid, rowg = i0 + row;
      float mo = sM[row], mx = mo;
      for (int j = 0; j < 64; ++j) {
        float s = (sS[row * SSTR + j] + sT[row * TSTR + (j + 31 - row)]) * 0.03125f;
        if (j0 + j > rowg + MEMLEN) s = -1.0e30f;  // causal mask
        sS[row * SSTR + j] = s;
        mx = fmaxf(mx, s);
      }
      float al = __expf(mo - mx);
      float l = sL[row] * al;
      for (int j = 0; j < 64; ++j) {
        float pp = __expf(sS[row * SSTR + j] - mx);
        l += pp;
        sP[row * 64 + j] = (__bf16)pp;
      }
      sM[row] = mx; sL[row] = l; sAl[row] = al;
    }
    __syncthreads();

    // rescale running O, then O += P * V
#pragma unroll
    for (int v = 0; v < 8; ++v) {
      float al = sAl[rs * 16 + v + 8 * hi];
      o[0][v] *= al; o[1][v] *= al;
    }
    v16bf ap[2];
#pragma unroll
    for (int kh = 0; kh < 2; ++kh)
      ap[kh] = load_frag_a(sP + rs * 16 * 64 + kh * 32, 64, lane);
#pragma unroll
    for (int cb = 0; cb < 2; ++cb)
#pragma unroll
      for (int kh = 0; kh < 2; ++kh)
        o[cb] = wmma_bf16(ap[kh],
                          load_frag_b(sVT + (ch * 32 + cb * 16) * 64 + kh * 32, 64, lane),
                          o[cb]);
  }
  __syncthreads();
#pragma unroll
  for (int cb = 0; cb < 2; ++cb)
#pragma unroll
    for (int v = 0; v < 8; ++v) {
      int grow = rs * 16 + v + 8 * hi;
      int d = ch * 32 + cb * 16 + n;
      float val = o[cb][v] / sL[grow];
      aout[((size_t)(b * QLEN + i0 + grow)) * EMBED + h * D_HEAD + d] = (__bf16)val;
    }
}

// ------------------------------------------------- output proj + residual
__global__ __launch_bounds__(128)
void out_gemm_kernel(const __bf16* __restrict__ a, const float* __restrict__ Wo,
                     const float* __restrict__ wres, float* __restrict__ x) {
  __shared__ __bf16 sA[64 * GSTR];
  __shared__ __bf16 sW[64 * GSTR];
  const int m0 = blockIdx.x * 64, n0 = blockIdx.y * 64;
  const int tid = threadIdx.x, wid = tid >> 5, lane = tid & 31;
  v8f acc[4];
#pragma unroll
  for (int cb = 0; cb < 4; ++cb) acc[cb] = vzero8();
  for (int k0 = 0; k0 < EMBED; k0 += 32) {
    for (int e = tid; e < 64 * 32; e += 128) {
      int rr_ = e >> 5, cc = e & 31;
      sA[rr_ * GSTR + cc] = a[(size_t)(m0 + rr_) * EMBED + k0 + cc];
    }
    for (int e = tid; e < 64 * 32; e += 128) {
      int kk = e >> 6, nn = e & 63;
      sW[nn * GSTR + kk] = (__bf16)Wo[(size_t)(k0 + kk) * EMBED + n0 + nn];
    }
    __syncthreads();
    v16bf av = load_frag_a(sA + wid * 16 * GSTR, GSTR, lane);
#pragma unroll
    for (int cb = 0; cb < 4; ++cb)
      acc[cb] = wmma_bf16(av, load_frag_b(sW + cb * 16 * GSTR, GSTR, lane), acc[cb]);
    __syncthreads();
  }
  const int n = lane & 15, hi = lane >> 4;
#pragma unroll
  for (int cb = 0; cb < 4; ++cb)
#pragma unroll
    for (int v = 0; v < 8; ++v) {
      int row = wid * 16 + v + 8 * hi, col = n0 + cb * 16 + n;
      size_t gi = (size_t)(m0 + row) * EMBED + col;
      x[gi] = acc[cb][v] + wres[gi];
    }
}

// ------------------------------------------------------------------ layernorm
__global__ __launch_bounds__(256)
void ln_kernel(const float* __restrict__ x, const float* __restrict__ g,
               const float* __restrict__ bta, float* __restrict__ out) {
  __shared__ float red[256];
  const int row = blockIdx.x, tid = threadIdx.x;
  const float* xr = x + (size_t)row * EMBED;
  float s = 0.f;
  for (int j = tid; j < EMBED; j += 256) s += xr[j];
  red[tid] = s; __syncthreads();
  for (int st = 128; st > 0; st >>= 1) {
    if (tid < st) red[tid] += red[tid + st];
    __syncthreads();
  }
  const float mu = red[0] * (1.0f / EMBED);
  __syncthreads();
  float v = 0.f;
  for (int j = tid; j < EMBED; j += 256) { float d = xr[j] - mu; v += d * d; }
  red[tid] = v; __syncthreads();
  for (int st = 128; st > 0; st >>= 1) {
    if (tid < st) red[tid] += red[tid + st];
    __syncthreads();
  }
  const float inv = rsqrtf(red[0] * (1.0f / EMBED) + 1e-3f);
  for (int j = tid; j < EMBED; j += 256)
    out[(size_t)row * EMBED + j] = (xr[j] - mu) * inv * g[j] + bta[j];
}

// ------------------------------------------------------------------- launcher
extern "C" void kernel_launch(void* const* d_in, const int* in_sizes, int n_in,
                              void* d_out, int out_size, void* d_ws, size_t ws_size,
                              hipStream_t stream) {
  (void)in_sizes; (void)n_in; (void)out_size; (void)ws_size;
  const float* w      = (const float*)d_in[0];
  const float* r      = (const float*)d_in[1];
  // d_in[2] attn_mask: reproduced analytically (j > i + MEM_LEN)
  const float* rwb    = (const float*)d_in[3];
  const float* rrb    = (const float*)d_in[4];
  const float* member = (const float*)d_in[5];
  const float* Wq     = (const float*)d_in[6];
  const float* Wk     = (const float*)d_in[7];
  const float* Wv     = (const float*)d_in[8];
  const float* Wr     = (const float*)d_in[9];
  const float* Wo     = (const float*)d_in[10];
  const float* gamma  = (const float*)d_in[11];
  const float* beta   = (const float*)d_in[12];

  char* p = (char*)d_ws;
  __bf16* qrw  = (__bf16*)p; p += (size_t)BATCH * N_HEAD * QLEN * D_HEAD * 2;
  __bf16* qrr  = (__bf16*)p; p += (size_t)BATCH * N_HEAD * QLEN * D_HEAD * 2;
  __bf16* kbf  = (__bf16*)p; p += (size_t)BATCH * N_HEAD * KLEN * D_HEAD * 2;
  __bf16* vbf  = (__bf16*)p; p += (size_t)BATCH * N_HEAD * KLEN * D_HEAD * 2;
  __bf16* rrbf = (__bf16*)p; p += (size_t)N_HEAD * KLEN * D_HEAD * 2;
  __bf16* aout = (__bf16*)p; p += (size_t)BATCH * QLEN * EMBED * 2;
  float*  xbuf = (float*)p;  p += (size_t)BATCH * QLEN * EMBED * 4;

  proj_q_kernel<<<dim3(32, 16), 128, 0, stream>>>(w, Wq, rwb, rrb, qrw, qrr);
  proj_cat_kernel<<<dim3(64, 16), 128, 0, stream>>>(member, w, Wk, kbf);
  proj_cat_kernel<<<dim3(64, 16), 128, 0, stream>>>(member, w, Wv, vbf);
  proj_r_kernel<<<dim3(32, 16), 128, 0, stream>>>(r, Wr, rrbf);
  attn_kernel<<<dim3(32, 16, 2), 128, 0, stream>>>(qrw, qrr, kbf, vbf, rrbf, aout);
  out_gemm_kernel<<<dim3(32, 16), 128, 0, stream>>>(aout, Wo, w, xbuf);
  ln_kernel<<<dim3(2048), 256, 0, stream>>>(xbuf, gamma, beta, (float*)d_out);
}
